// DynamicGNNdec2_30751965839949
// MI455X (gfx1250) — matrix operand order repaired and checked by reading, not compiled, not run on hardware
//
#include <hip/hip_runtime.h>
#include <hip/hip_bf16.h>
#include <cstdint>
#include <cstddef>

typedef __attribute__((ext_vector_type(16))) _Float16 v16h;
typedef __attribute__((ext_vector_type(8)))  float    v8f;
typedef __attribute__((ext_vector_type(4)))  unsigned int v4u;
typedef __attribute__((ext_vector_type(8)))  int      v8i_;
typedef __attribute__((ext_vector_type(4)))  int      v4i_;

#define DEV_INLINE __device__ __forceinline__

namespace {

constexpr int B_  = 2048;
constexpr int N0_ = 39, F0_ = 39, H_ = 64, K_ = 10;
constexpr int N1_ = 32, N2_ = 26, N3_ = 21;

// ---------------- LDS float max via CAS (deterministic: max is commutative) ----
DEV_INLINE void lds_fmax(float* p, float v) {
  unsigned int* up = (unsigned int*)p;
  unsigned int cur = *up;
  while (v > __uint_as_float(cur)) {
    unsigned int prev = atomicCAS(up, cur, __float_as_uint(v));
    if (prev == cur) break;
    cur = prev;
  }
}

// ---------------- Tensor Data Mover: 1-D f32 tile, global -> LDS --------------
// D# per CDNA5 ISA ch.8: group0 = {count=1, lds_addr, global_addr[56:0], type=2}
// group1 = {data_size=4B, tensor_dim0=nelem, tensor_dim1=1, tile_dim0=nelem,
//           tile_dim1=1, dim0_stride=nelem}; groups 2/3 unused (<=2D).
// Issue from ONE wave only (TDM ignores EXEC: one instruction == one DMA);
// completion tracked on TENSORcnt.
DEV_INLINE void tdm_load_linear_f32(unsigned lds_off, const void* gp, int nelem) {
  unsigned long long ga = (unsigned long long)(uintptr_t)gp;
  v4u g0;
  g0.x = 1u;                                           // count=1, user mode
  g0.y = lds_off;                                      // lds_addr (bytes)
  g0.z = (unsigned)(ga & 0xffffffffu);                 // global_addr[31:0]
  g0.w = (unsigned)((ga >> 32) & 0x01ffffffu)          // global_addr[56:32]
       | 0x80000000u;                                  // type=2 ("image")
  v8i_ g1 = {};
  unsigned ne = (unsigned)nelem;
  g1[0] = (int)(2u << 16);                             // data_size = 4 bytes
  g1[1] = (int)((ne & 0xffffu) << 16);                 // tensor_dim0 lo16
  g1[2] = (int)((ne >> 16) | (1u << 16));              // tensor_dim0 hi16, tensor_dim1=1
  g1[3] = (int)((ne & 0xffffu) << 16);                 // tile_dim0 (<=65535)
  g1[4] = 1;                                           // tile_dim1 = 1, tile_dim2 = 0
  g1[5] = (int)ne;                                     // tensor_dim0_stride lo32
  g1[6] = 0;
  g1[7] = 0;
  v4i_ z4 = {0, 0, 0, 0};
#if __has_builtin(__builtin_amdgcn_tensor_load_to_lds)
#if __clang_major__ >= 23
  v8i_ z8 = {};
  __builtin_amdgcn_tensor_load_to_lds(g0, g1, z4, z4, z8, 0);
#else
  __builtin_amdgcn_tensor_load_to_lds(g0, g1, z4, z4, 0);
#endif
#else
  (void)g0; (void)g1; (void)z4;
#endif
}

// ---------------- wave-level 16x16 WMMA tile: D = A[m0:,:] * Wt^T[:,n0:] ------
// As: LDS row-major f16 [rows][KPAD]; Wt: LDS f16 [N][KPAD] (i.e. W transposed).
// A frag (16x32 f16): lanes 0-15 hold K 0..15, lanes 16-31 hold K 16..31
// B frag (32x16 f16): lanes 0-15 hold K 0..15, lanes 16-31 hold K 16..31
// C/D (16x16 f32): VGPR v -> M = v + 8*(lane>=16), N = lane&15
template<int KPAD>
DEV_INLINE v8f wmma_tile(const _Float16* As, const _Float16* Wt,
                         int m0, int n0, int lane) {
  v8f c = {};
  const int mrow  = m0 + (lane & 15);
  const int ncol  = n0 + (lane & 15);
  const int abase = (lane >> 4) * 16;
  const int bbase = (lane >> 4) * 16;
  for (int ks = 0; ks < KPAD; ks += 32) {
    v16h a, b;
#pragma unroll
    for (int e = 0; e < 8; ++e) {
      a[e]     = As[mrow * KPAD + ks + abase + e];
      a[e + 8] = As[mrow * KPAD + ks + abase + e + 8];
    }
#pragma unroll
    for (int e = 0; e < 16; ++e)
      b[e] = Wt[ncol * KPAD + ks + bbase + e];
    c = __builtin_amdgcn_wmma_f32_16x16x32_f16(false, a, false, b,
                                               (short)0, c, false, false);
  }
  return c;
}

// ---------------- DynamicEdgeConv: kNN + mlp([xi, xj-xi]) + max-agg ----------
// hout = relu(max_k(feat @ W) + bias)  (relu/bias commute with max over k)
template<int NN, int FIN>
__global__ void __launch_bounds__(128)
edgeconv_kernel(const float* __restrict__ xg,   // [B, NN, FIN]
                const float* __restrict__ w,    // [2*FIN, 64]
                const float* __restrict__ bias, // [64]
                float* __restrict__ hout) {     // [B, NN, 64]
  constexpr int KDIM = 2 * FIN;
  constexpr int KPAD = (KDIM + 31) & ~31;
  constexpr int RR   = NN * K_;
  constexpr int RPAD = (RR + 15) & ~15;

  __shared__ float    xs[NN][FIN];
  __shared__ _Float16 Wt[H_][KPAD];
  __shared__ float    distm[NN][NN];
  __shared__ int      nbr[NN][K_];
  __shared__ float    acc[NN][H_];
  extern __shared__ _Float16 ef[];              // [RPAD][KPAD]

  const int b = blockIdx.x, tid = threadIdx.x;
  const int lane = tid & 31, wave = tid >> 5;
  const float* x = xg + (size_t)b * NN * FIN;

  // Stage x tile and raw weights into LDS. Raw W is overlaid on the ef region
  // (consumed into Wt before ef is built; barriers order the reuse).
  float* Wraw = (float*)ef;
#if __has_builtin(__builtin_amdgcn_tensor_load_to_lds) && \
    __has_builtin(__builtin_amdgcn_s_wait_tensorcnt)
  if (wave == 0) {
    tdm_load_linear_f32((unsigned)(uintptr_t)&xs[0][0], x, NN * FIN);
    tdm_load_linear_f32((unsigned)(uintptr_t)Wraw, w, KDIM * H_);
    __builtin_amdgcn_s_wait_tensorcnt(0);
  }
  __syncthreads();
#else
  for (int t = tid; t < NN * FIN; t += 128) (&xs[0][0])[t] = x[t];
  for (int t = tid; t < KDIM * H_; t += 128) Wraw[t] = w[t];
  __syncthreads();
#endif

  // transpose + f16-convert weights out of the staging overlay
  for (int t = tid; t < H_ * KPAD; t += 128) {
    int n = t / KPAD, k = t % KPAD;
    Wt[n][k] = (k < KDIM) ? (_Float16)Wraw[k * H_ + n] : (_Float16)0.0f;
  }

  // pairwise squared distances (self-distance exactly 0 -> self included in kNN)
  for (int t = tid; t < NN * NN; t += 128) {
    int i = t / NN, j = t % NN;
    float s = 0.0f;
    for (int f = 0; f < FIN; ++f) { float d = xs[i][f] - xs[j][f]; s += d * d; }
    distm[i][j] = s;
  }
  __syncthreads();

  // top-K nearest: iterative argmin, strict < -> lowest index on ties (== top_k)
  for (int i = tid; i < NN; i += 128) {
    for (int k = 0; k < K_; ++k) {
      float best = 3.4e38f; int bj = 0;
      for (int j = 0; j < NN; ++j) {
        float dv = distm[i][j];
        if (dv < best) { best = dv; bj = j; }
      }
      nbr[i][k] = bj;
      distm[i][bj] = 3.4e38f;
    }
  }
  __syncthreads();   // Wt conversion reads of Wraw done before ef overwrite

  // edge features [xi, xj - xi], f16, zero-padded to RPAD x KPAD
  for (int t = tid; t < RPAD * KPAD; t += 128) {
    int r = t / KPAD, c = t % KPAD;
    _Float16 v = (_Float16)0.0f;
    if (r < RR && c < KDIM) {
      int i = r / K_, j = nbr[i][r % K_];
      v = (c < FIN) ? (_Float16)xs[i][c]
                    : (_Float16)(xs[j][c - FIN] - xs[i][c - FIN]);
    }
    ef[(size_t)r * KPAD + c] = v;
  }
  for (int t = tid; t < NN * H_; t += 128) (&acc[0][0])[t] = -3.0e38f;
  __syncthreads();

  // GEMM: wave owns a 16-col slab; max-scatter rows -> node accumulator
  const int n0 = wave * 16;
  for (int m0 = 0; m0 < RPAD; m0 += 16) {
    v8f c = wmma_tile<KPAD>(ef, &Wt[0][0], m0, n0, lane);
    const int ncol = n0 + (lane & 15);
    const int rb   = m0 + ((lane >> 4) << 3);
#pragma unroll
    for (int v = 0; v < 8; ++v) {
      int r = rb + v;
      if (r < RR) lds_fmax(&acc[r / K_][ncol], c[v]);
    }
  }
  __syncthreads();

  float* ho = hout + (size_t)b * NN * H_;
  for (int t = tid; t < NN * H_; t += 128) {
    float v = (&acc[0][0])[t] + bias[t & (H_ - 1)];
    ho[t] = v > 0.0f ? v : 0.0f;
  }
}

// ---------------- BatchNorm batch statistics (training mode) ------------------
// y = x*scale + shift ; scale = gamma*rsqrt(var+eps), shift = beta - mu*scale
__global__ void __launch_bounds__(256)
bnstats_kernel(const float* __restrict__ h, int rows,
               const float* __restrict__ gamma, const float* __restrict__ beta,
               float* __restrict__ scale, float* __restrict__ shift) {
  const int c = blockIdx.x;                    // 64 channels
  __shared__ float ssum[256], ssq[256];
  float s = 0.f, q = 0.f;
  for (int r = threadIdx.x; r < rows; r += 256) {
    float v = h[(size_t)r * H_ + c];
    s += v; q += v * v;
  }
  ssum[threadIdx.x] = s; ssq[threadIdx.x] = q;
  __syncthreads();
  for (int st = 128; st > 0; st >>= 1) {
    if (threadIdx.x < st) {
      ssum[threadIdx.x] += ssum[threadIdx.x + st];
      ssq[threadIdx.x]  += ssq[threadIdx.x + st];
    }
    __syncthreads();
  }
  if (threadIdx.x == 0) {
    float mu  = ssum[0] / rows;
    float var = ssq[0] / rows - mu * mu;
    float sc  = rsqrtf(var + 1e-5f) * gamma[c];
    scale[c] = sc;
    shift[c] = beta[c] - mu * sc;
  }
}

// ---------------- BN+relu, GCN score, SAG top-k pool --------------------------
template<int NN, int NKEEP, bool STAGE1>
__global__ void __launch_bounds__(128)
pool_kernel(const float* __restrict__ h,       // [B, NN, 64] pre-BN
            const float* __restrict__ scale, const float* __restrict__ shift,
            const float* __restrict__ adj_in,  // [B, NN, NN]
            const float* __restrict__ sw, const float* __restrict__ sb,
            float* __restrict__ xout,          // [B, NKEEP, 64]
            float* __restrict__ adjout,        // [B, NKEEP, NKEEP]
            int*   __restrict__ idxout,        // [B, NKEEP]
            _Float16* __restrict__ xscat) {    // [B, NN, 64]  (STAGE1 only)
  __shared__ float hb[NN][H_];
  __shared__ float adjs[NN][NN];
  __shared__ float sc0[NN], sc1[NN], dinv[NN];
  __shared__ int   idx[NKEEP];
  __shared__ float tnh[NKEEP];
  __shared__ int   slot[NN];

  const int b = blockIdx.x, tid = threadIdx.x;
  const float* hg = h + (size_t)b * NN * H_;
  const float* ag = adj_in + (size_t)b * NN * NN;
  for (int t = tid; t < NN * H_; t += 128) {
    int n = t & (H_ - 1);
    float v = hg[t] * scale[n] + shift[n];
    (&hb[0][0])[t] = v > 0.f ? v : 0.f;       // relu(bn(h))
  }
  for (int t = tid; t < NN * NN; t += 128) (&adjs[0][0])[t] = ag[t];
  __syncthreads();

  // GCN score with self-loops + symmetric norm
  for (int i = tid; i < NN; i += 128) {
    float s = 1.0f;
    for (int j = 0; j < NN; ++j) s += adjs[i][j];
    dinv[i] = rsqrtf(s);
    float d = 0.f;
    for (int n = 0; n < H_; ++n) d += hb[i][n] * sw[n];
    sc0[i] = d * dinv[i];
  }
  __syncthreads();
  for (int i = tid; i < NN; i += 128) {
    float s = sc0[i];                          // self-loop term
    for (int j = 0; j < NN; ++j) s += adjs[i][j] * sc0[j];
    sc1[i] = s * dinv[i] + sb[0];
  }
  __syncthreads();
  if (tid == 0) {                              // top-NKEEP, strict > == top_k ties
    for (int k = 0; k < NKEEP; ++k) {
      float best = -3.4e38f; int bi = 0;
      for (int j = 0; j < NN; ++j) {
        float v = sc1[j];
        if (v > best) { best = v; bi = j; }
      }
      idx[k] = bi; tnh[k] = tanhf(best); sc1[bi] = -3.4e38f;
    }
  }
  __syncthreads();

  float* xo = xout + (size_t)b * NKEEP * H_;
  for (int t = tid; t < NKEEP * H_; t += 128) {
    int s = t >> 6, n = t & 63;
    xo[t] = hb[idx[s]][n] * tnh[s];
  }
  float* ao = adjout + (size_t)b * NKEEP * NKEEP;
  for (int t = tid; t < NKEEP * NKEEP; t += 128) {
    int r = t / NKEEP, c = t % NKEEP;
    ao[t] = adjs[idx[r]][idx[c]];
  }
  int* io = idxout + (size_t)b * NKEEP;
  for (int t = tid; t < NKEEP; t += 128) io[t] = idx[t];

  if constexpr (STAGE1) {                      // x_out scatter (zeros elsewhere)
    for (int t = tid; t < NN; t += 128) slot[t] = -1;
    __syncthreads();
    for (int t = tid; t < NKEEP; t += 128) slot[idx[t]] = t;
    __syncthreads();
    _Float16* xs2 = xscat + (size_t)b * NN * H_;
    for (int t = tid; t < NN * H_; t += 128) {
      int n = t >> 6, c = t & 63;
      int s = slot[n];
      xs2[t] = (s < 0) ? (_Float16)0.f : (_Float16)(hb[idx[s]][c] * tnh[s]);
    }
  }
}

// ---------------- degree MLP: relu(relu(relu(x@w4+b4)@w5+b5)@w6+b6) -----------
__global__ void __launch_bounds__(128)
degree_kernel(const _Float16* __restrict__ xin,   // [B, 39, 64]
              const float* __restrict__ w4, const float* __restrict__ b4,
              const float* __restrict__ w5, const float* __restrict__ b5,
              const float* __restrict__ w6, const float* __restrict__ b6,
              float* __restrict__ xdeg) {          // [B, 39]
  constexpr int RP = 48;
  __shared__ _Float16 A1[RP][H_];
  __shared__ _Float16 W4t[H_][H_];
  __shared__ _Float16 A2[RP][H_];
  __shared__ _Float16 W5t[32][H_];
  __shared__ float    O2[RP][32];
  const int b = blockIdx.x, tid = threadIdx.x, lane = tid & 31, wave = tid >> 5;
  const _Float16* xg = xin + (size_t)b * N0_ * H_;
  for (int t = tid; t < RP * H_; t += 128) {
    int r = t >> 6, c = t & 63;
    A1[r][c] = (r < N0_) ? xg[r * H_ + c] : (_Float16)0.f;
  }
  for (int t = tid; t < H_ * H_; t += 128) { int n = t >> 6, k = t & 63; W4t[n][k] = (_Float16)w4[k * H_ + n]; }
  for (int t = tid; t < 32 * H_; t += 128)  { int n = t >> 6, k = t & 63; W5t[n][k] = (_Float16)w5[k * 32 + n]; }
  __syncthreads();
  for (int job = wave; job < 12; job += 4) {           // 3 m-tiles x 4 n-tiles
    int m0 = (job >> 2) * 16, n0 = (job & 3) * 16;
    v8f c = wmma_tile<H_>(&A1[0][0], &W4t[0][0], m0, n0, lane);
    int ncol = n0 + (lane & 15), rb = m0 + ((lane >> 4) << 3);
#pragma unroll
    for (int v = 0; v < 8; ++v) {
      float y = c[v] + b4[ncol];
      A2[rb + v][ncol] = (_Float16)(y > 0.f ? y : 0.f);
    }
  }
  __syncthreads();
  for (int job = wave; job < 6; job += 4) {            // 3 m-tiles x 2 n-tiles
    int m0 = (job >> 1) * 16, n0 = (job & 1) * 16;
    v8f c = wmma_tile<H_>(&A2[0][0], &W5t[0][0], m0, n0, lane);
    int ncol = n0 + (lane & 15), rb = m0 + ((lane >> 4) << 3);
#pragma unroll
    for (int v = 0; v < 8; ++v) {
      float y = c[v] + b5[ncol];
      O2[rb + v][ncol] = y > 0.f ? y : 0.f;
    }
  }
  __syncthreads();
  for (int r = tid; r < N0_; r += 128) {
    float s = b6[0];
    for (int k = 0; k < 32; ++k) s += O2[r][k] * w6[k];
    xdeg[(size_t)b * N0_ + r] = s > 0.f ? s : 0.f;
  }
}

// ---------------- classifier head: relu(relu(x@l1)@l2)@l3 ---------------------
__global__ void __launch_bounds__(128)
head_kernel(const float* __restrict__ x3,            // [B, 21, 64]
            const float* __restrict__ w1, const float* __restrict__ b1,
            const float* __restrict__ w2, const float* __restrict__ b2,
            const float* __restrict__ w3, const float* __restrict__ b3,
            float* __restrict__ z) {                  // [B*21, 2]
  constexpr int RP = 32;
  __shared__ _Float16 A1[RP][H_];
  __shared__ _Float16 W1t[H_][H_];
  __shared__ _Float16 A2[RP][H_];
  __shared__ _Float16 W2t[32][H_];
  __shared__ float    O2[RP][32];
  const int b = blockIdx.x, tid = threadIdx.x, lane = tid & 31, wave = tid >> 5;
  const float* xg = x3 + (size_t)b * N3_ * H_;
  for (int t = tid; t < RP * H_; t += 128) {
    int r = t >> 6, c = t & 63;
    A1[r][c] = (r < N3_) ? (_Float16)xg[r * H_ + c] : (_Float16)0.f;
  }
  for (int t = tid; t < H_ * H_; t += 128) { int n = t >> 6, k = t & 63; W1t[n][k] = (_Float16)w1[k * H_ + n]; }
  for (int t = tid; t < 32 * H_; t += 128)  { int n = t >> 6, k = t & 63; W2t[n][k] = (_Float16)w2[k * 32 + n]; }
  __syncthreads();
  for (int job = wave; job < 8; job += 4) {            // 2 m-tiles x 4 n-tiles
    int m0 = (job & 1) * 16, n0 = (job >> 1) * 16;
    v8f c = wmma_tile<H_>(&A1[0][0], &W1t[0][0], m0, n0, lane);
    int ncol = n0 + (lane & 15), rb = m0 + ((lane >> 4) << 3);
#pragma unroll
    for (int v = 0; v < 8; ++v) {
      float y = c[v] + b1[ncol];
      A2[rb + v][ncol] = (_Float16)(y > 0.f ? y : 0.f);
    }
  }
  __syncthreads();
  for (int job = wave; job < 4; job += 4) {            // 2 m-tiles x 2 n-tiles
    int m0 = (job & 1) * 16, n0 = (job >> 1) * 16;
    v8f c = wmma_tile<H_>(&A2[0][0], &W2t[0][0], m0, n0, lane);
    int ncol = n0 + (lane & 15), rb = m0 + ((lane >> 4) << 3);
#pragma unroll
    for (int v = 0; v < 8; ++v) {
      float y = c[v] + b2[ncol];
      O2[rb + v][ncol] = y > 0.f ? y : 0.f;
    }
  }
  __syncthreads();
  for (int r = tid; r < N3_; r += 128) {
    float z0 = b3[0], z1 = b3[1];
    for (int k = 0; k < 32; ++k) {
      z0 += O2[r][k] * w3[k * 2 + 0];
      z1 += O2[r][k] * w3[k * 2 + 1];
    }
    size_t o = ((size_t)b * N3_ + r) * 2;
    z[o] = z0; z[o + 1] = z1;                          // no relu on final layer
  }
}

// ---------------- degree(edge_index[0]) = rowsum(adj), flat [B*N0] ------------
__global__ void __launch_bounds__(256)
degsum_kernel(const float* __restrict__ adj, float* __restrict__ deg) {
  int t = blockIdx.x * 256 + threadIdx.x;
  if (t >= B_ * N0_) return;
  const float* a = adj + (size_t)t * N0_;
  float s = 0.f;
  for (int j = 0; j < N0_; ++j) s += a[j];
  deg[t] = s;
}

// ---------------- gt/pr gathers (faithful flat-index reuse) -------------------
__global__ void __launch_bounds__(256)
gather_kernel(const int* __restrict__ idx, int nkeep, int stride,
              const float* __restrict__ deg, const float* __restrict__ xdeg,
              float* __restrict__ gt, float* __restrict__ pr, int total) {
  int t = blockIdx.x * 256 + threadIdx.x;
  if (t >= total) return;
  int b = t / nkeep;
  long p = (long)b * stride + idx[t];   // indexes the FULL B*N0 arrays (as ref)
  gt[t] = deg[p];
  pr[t] = xdeg[p];
}

constexpr int kpad_of(int fin) { return (2 * fin + 31) & ~31; }
constexpr int rpad_of(int nn)  { return (nn * K_ + 15) & ~15; }

} // namespace

extern "C" void kernel_launch(void* const* d_in, const int* in_sizes, int n_in,
                              void* d_out, int out_size, void* d_ws, size_t ws_size,
                              hipStream_t stream) {
  (void)in_sizes; (void)n_in; (void)out_size; (void)ws_size;
  // params flattened in sorted-dict order: bn1..bn3, lin1..lin6, mlp1..mlp3, score1..3
  const float* x     = (const float*)d_in[0];
  const float* adj   = (const float*)d_in[1];
  const float* bn1g  = (const float*)d_in[2];  const float* bn1b  = (const float*)d_in[3];
  const float* bn2g  = (const float*)d_in[4];  const float* bn2b  = (const float*)d_in[5];
  const float* bn3g  = (const float*)d_in[6];  const float* bn3b  = (const float*)d_in[7];
  const float* lin1w = (const float*)d_in[8];  const float* lin1b = (const float*)d_in[9];
  const float* lin2w = (const float*)d_in[10]; const float* lin2b = (const float*)d_in[11];
  const float* lin3w = (const float*)d_in[12]; const float* lin3b = (const float*)d_in[13];
  const float* lin4w = (const float*)d_in[14]; const float* lin4b = (const float*)d_in[15];
  const float* lin5w = (const float*)d_in[16]; const float* lin5b = (const float*)d_in[17];
  const float* lin6w = (const float*)d_in[18]; const float* lin6b = (const float*)d_in[19];
  const float* mlp1w = (const float*)d_in[20]; const float* mlp1b = (const float*)d_in[21];
  const float* mlp2w = (const float*)d_in[22]; const float* mlp2b = (const float*)d_in[23];
  const float* mlp3w = (const float*)d_in[24]; const float* mlp3b = (const float*)d_in[25];
  const float* sc1w  = (const float*)d_in[26]; const float* sc1b  = (const float*)d_in[27];
  const float* sc2w  = (const float*)d_in[28]; const float* sc2b  = (const float*)d_in[29];
  const float* sc3w  = (const float*)d_in[30]; const float* sc3b  = (const float*)d_in[31];

  // workspace bump allocator
  char* p = (char*)d_ws;
  auto alloc = [&](size_t bytes) -> char* {
    char* r = p; p += (bytes + 255) & ~(size_t)255; return r;
  };
  float*    h1    = (float*)alloc((size_t)B_ * N0_ * H_ * 4);
  float*    h2    = (float*)alloc((size_t)B_ * N1_ * H_ * 4);
  float*    h3    = (float*)alloc((size_t)B_ * N2_ * H_ * 4);
  float*    x1p   = (float*)alloc((size_t)B_ * N1_ * H_ * 4);
  float*    x2p   = (float*)alloc((size_t)B_ * N2_ * H_ * 4);
  float*    x3p   = (float*)alloc((size_t)B_ * N3_ * H_ * 4);
  float*    adj1  = (float*)alloc((size_t)B_ * N1_ * N1_ * 4);
  float*    adj2  = (float*)alloc((size_t)B_ * N2_ * N2_ * 4);
  float*    adj3  = (float*)alloc((size_t)B_ * N3_ * N3_ * 4);
  int*      idx1  = (int*)alloc((size_t)B_ * N1_ * 4);
  int*      idx2  = (int*)alloc((size_t)B_ * N2_ * 4);
  int*      idx3  = (int*)alloc((size_t)B_ * N3_ * 4);
  _Float16* xscat = (_Float16*)alloc((size_t)B_ * N0_ * H_ * 2);
  float*    xdeg  = (float*)alloc((size_t)B_ * N0_ * 4);
  float*    deg   = (float*)alloc((size_t)B_ * N0_ * 4);
  float*    bnsc1 = (float*)alloc(64 * 4);  float* bnsh1 = (float*)alloc(64 * 4);
  float*    bnsc2 = (float*)alloc(64 * 4);  float* bnsh2 = (float*)alloc(64 * 4);
  float*    bnsc3 = (float*)alloc(64 * 4);  float* bnsh3 = (float*)alloc(64 * 4);

  // output regions (fp32), concatenated in return order
  float* out = (float*)d_out;
  float* z   = out;                       // 2048*21*2 = 86016
  float* gt1 = out + 86016;               // 65536
  float* pr1 = out + 151552;              // 65536
  float* gt2 = out + 217088;              // 53248
  float* pr2 = out + 270336;              // 53248
  float* gt3 = out + 323584;              // 43008
  float* pr3 = out + 366592;              // 43008

  degsum_kernel<<<(B_ * N0_ + 255) / 256, 256, 0, stream>>>(adj, deg);

  // ---- block 1 ----
  edgeconv_kernel<N0_, F0_><<<B_, 128, rpad_of(N0_) * kpad_of(F0_) * 2, stream>>>(
      x, mlp1w, mlp1b, h1);
  bnstats_kernel<<<64, 256, 0, stream>>>(h1, B_ * N0_, bn1g, bn1b, bnsc1, bnsh1);
  pool_kernel<N0_, N1_, true><<<B_, 128, 0, stream>>>(
      h1, bnsc1, bnsh1, adj, sc1w, sc1b, x1p, adj1, idx1, xscat);
  degree_kernel<<<B_, 128, 0, stream>>>(xscat, lin4w, lin4b, lin5w, lin5b,
                                        lin6w, lin6b, xdeg);
  gather_kernel<<<(B_ * N1_ + 255) / 256, 256, 0, stream>>>(
      idx1, N1_, N0_, deg, xdeg, gt1, pr1, B_ * N1_);

  // ---- block 2 ----
  edgeconv_kernel<N1_, H_><<<B_, 128, rpad_of(N1_) * kpad_of(H_) * 2, stream>>>(
      x1p, mlp2w, mlp2b, h2);
  bnstats_kernel<<<64, 256, 0, stream>>>(h2, B_ * N1_, bn2g, bn2b, bnsc2, bnsh2);
  pool_kernel<N1_, N2_, false><<<B_, 128, 0, stream>>>(
      h2, bnsc2, bnsh2, adj1, sc2w, sc2b, x2p, adj2, idx2, nullptr);
  gather_kernel<<<(B_ * N2_ + 255) / 256, 256, 0, stream>>>(
      idx2, N2_, N1_, deg, xdeg, gt2, pr2, B_ * N2_);

  // ---- block 3 ----
  edgeconv_kernel<N2_, H_><<<B_, 128, rpad_of(N2_) * kpad_of(H_) * 2, stream>>>(
      x2p, mlp3w, mlp3b, h3);
  bnstats_kernel<<<64, 256, 0, stream>>>(h3, B_ * N2_, bn3g, bn3b, bnsc3, bnsh3);
  pool_kernel<N2_, N3_, false><<<B_, 128, 0, stream>>>(
      h3, bnsc3, bnsh3, adj2, sc3w, sc3b, x3p, adj3, idx3, nullptr);
  gather_kernel<<<(B_ * N3_ + 255) / 256, 256, 0, stream>>>(
      idx3, N3_, N2_, deg, xdeg, gt3, pr3, B_ * N3_);

  // ---- classifier head ----
  head_kernel<<<B_, 128, 0, stream>>>(x3p, lin1w, lin1b, lin2w, lin2b,
                                      lin3w, lin3b, z);
}
